// SelfAttention_33131377722136
// MI455X (gfx1250) — matrix operand order repaired and checked by reading, not compiled, not run on hardware
//
#include <hip/hip_runtime.h>
#include <hip/hip_bf16.h>
#include <math.h>

// Problem constants (from reference): x[4,2048,512], 8 heads, head_dim 64.
#define BATCH 4
#define SEQ   2048
#define DMODEL 512
#define NHEAD 8
#define HDIM  64
#define NTOK  (BATCH * SEQ)   // 8192 tokens

typedef __attribute__((ext_vector_type(16))) __bf16 v16bf;
typedef __attribute__((ext_vector_type(8)))  __bf16 v8bf;
typedef __attribute__((ext_vector_type(8)))  float  v8f;
typedef __attribute__((ext_vector_type(4)))  float  v4f;
typedef __attribute__((ext_vector_type(4)))  int    v4i;

// CDNA5 async global->LDS path (gfx1250). Guarded so absence degrades to a
// synchronous LDS copy instead of breaking the build.
#if defined(__has_builtin)
#if __has_builtin(__builtin_amdgcn_global_load_async_to_lds_b128)
#define HAVE_ASYNC_LDS 1
#endif
#endif

static __device__ __forceinline__ void wait_asynccnt0() {
#if defined(__has_builtin) && __has_builtin(__builtin_amdgcn_s_wait_asynccnt)
  __builtin_amdgcn_s_wait_asynccnt(0);
#else
  asm volatile("s_wait_asynccnt 0" ::: "memory");
#endif
}

// ---------------------------------------------------------------------------
// WMMA fragment helpers, per CDNA5 ISA 16-bit VGPR layouts (CDNA5 ISA)
// A (16x32, row-major source): lane L holds row M=L%16.
//   lanes 0-15 : elems 0..7 = K 0..7,  elems 8..15 = K 16..23
//   lanes 16-31: elems 0..7 = K 8..15, elems 8..15 = K 24..31
// ---------------------------------------------------------------------------
static __device__ __forceinline__ v16bf load_frag_a(const __bf16* base, int ld) {
  const int lane = threadIdx.x & 31;
  const int row  = lane & 15;
  const int k0   = (lane < 16) ? 0 : 8;
  const __bf16* p = base + row * ld + k0;
  v8bf lo = *(const v8bf*)(p);          // K = k0 .. k0+7
  v8bf hi = *(const v8bf*)(p + 16);     // K = k0+16 .. k0+23
  return __builtin_shufflevector(lo, hi, 0,1,2,3,4,5,6,7,8,9,10,11,12,13,14,15);
}

// B (32x16): lane L holds column N=L%16; lanes 0-15 K=0..7,16..23, lanes 16-31 K=8..15,24..31
// (same K order as A; source: column n contiguous along K, stride ld).
static __device__ __forceinline__ v16bf load_frag_b(const __bf16* colbase, int ld) {
  const int lane = threadIdx.x & 31;
  const int n    = lane & 15;
  const int koff = (lane < 16) ? 0 : 8;
  const __bf16* p = colbase + (size_t)n * ld + koff;
  v8bf lo = *(const v8bf*)(p);
  v8bf hi = *(const v8bf*)(p + 16);
  return __builtin_shufflevector(lo, hi, 0,1,2,3,4,5,6,7,8,9,10,11,12,13,14,15);
}

static __device__ __forceinline__ v8f wmma_bf16(v16bf a, v16bf b, v8f c) {
  // (neg_a, A, neg_b, B, c_mod, C, reuse_a, reuse_b)
  return __builtin_amdgcn_wmma_f32_16x16x32_bf16(false, a, false, b, (short)0, c,
                                                 false, false);
}

// 16-byte global->LDS copy: async DMA on CDNA5, sync fallback otherwise.
static __device__ __forceinline__ void copy16_g2l(const __bf16* g, __bf16* l) {
#ifdef HAVE_ASYNC_LDS
  typedef __attribute__((address_space(1))) v4i g_v4i;  // global-side vec4 i32
  typedef __attribute__((address_space(3))) v4i l_v4i;  // LDS-side vec4 i32
  __builtin_amdgcn_global_load_async_to_lds_b128((g_v4i*)(g), (l_v4i*)(l), 0, 0);
#else
  *(v8bf*)l = *(const v8bf*)g;
#endif
}

// ---------------------------------------------------------------------------
// Kernel 1: fp32 -> bf16 elementwise convert (weights)
// ---------------------------------------------------------------------------
__global__ void cvt_bf16_kernel(const float* __restrict__ in, __bf16* __restrict__ out,
                                int n) {
  int i = blockIdx.x * blockDim.x + threadIdx.x;
  if (i < n) out[i] = (__bf16)in[i];
}

// ---------------------------------------------------------------------------
// Kernel 2: LayerNorm. One wave32 per token (512 elems -> 16 per lane).
// ---------------------------------------------------------------------------
__global__ __launch_bounds__(256) void ln_kernel(const float* __restrict__ x,
                                                 const float* __restrict__ gamma,
                                                 const float* __restrict__ beta,
                                                 float* __restrict__ xn,
                                                 __bf16* __restrict__ xnbf) {
  const int wave = threadIdx.x >> 5;
  const int lane = threadIdx.x & 31;
  const int t = blockIdx.x * 8 + wave;
  const float* row = x + (size_t)t * DMODEL;
  const int c0 = lane * 16;

  float vals[16];
  v4f* vp = (v4f*)vals;
#pragma unroll
  for (int i = 0; i < 4; ++i) vp[i] = *(const v4f*)(row + c0 + i * 4);

  float s = 0.f, s2 = 0.f;
#pragma unroll
  for (int i = 0; i < 16; ++i) { s += vals[i]; s2 += vals[i] * vals[i]; }
#pragma unroll
  for (int m = 1; m <= 16; m <<= 1) {
    s  += __shfl_xor(s,  m, 32);
    s2 += __shfl_xor(s2, m, 32);
  }
  const float mu   = s * (1.0f / DMODEL);
  const float var  = s2 * (1.0f / DMODEL) - mu * mu;
  const float rstd = rsqrtf(var + 1e-5f);

#pragma unroll
  for (int i = 0; i < 16; ++i) {
    float v = (vals[i] - mu) * rstd * gamma[c0 + i] + beta[c0 + i];
    xn[(size_t)t * DMODEL + c0 + i]   = v;
    xnbf[(size_t)t * DMODEL + c0 + i] = (__bf16)v;
  }
}

// ---------------------------------------------------------------------------
// Kernel 3: fused Q/K/V projection. C[t,o] = sum_k xn[t,k] * W[o,k].
// One wave per 16x64 output strip (4 accumulators). All 5 fragment loads of a
// K-step are issued before the 4 WMMAs so the scheduler can overlap WMMA
// execution with outstanding loads (partial s_wait_loadcnt instead of drains).
// ---------------------------------------------------------------------------
__global__ __launch_bounds__(256) void qkv_kernel(const __bf16* __restrict__ xnbf,
                                                  const __bf16* __restrict__ Wq,
                                                  const __bf16* __restrict__ Wk,
                                                  const __bf16* __restrict__ Wv,
                                                  __bf16* __restrict__ Qb,
                                                  __bf16* __restrict__ Kb,
                                                  __bf16* __restrict__ Vt) {
  const int wg   = blockIdx.x * 8 + (threadIdx.x >> 5);
  const int lane = threadIdx.x & 31;
  const int mat  = wg / (512 * 8);    // 0=Q 1=K 2=V
  const int rem  = wg % (512 * 8);
  const int tm   = rem >> 3;          // 0..511 (token tiles of 16)
  const int tn8  = rem & 7;           // 0..7   (64-wide output strips)

  const __bf16* W = (mat == 0) ? Wq : (mat == 1) ? Wk : Wv;
  const __bf16* abase = xnbf + (size_t)tm * 16 * DMODEL;
  const __bf16* bbase = W    + (size_t)tn8 * 64 * DMODEL;  // column o = row o of W

  v8f c0 = {}, c1 = {}, c2 = {}, c3 = {};
#pragma unroll 2
  for (int kb = 0; kb < DMODEL; kb += 32) {
    // issue every load of this K-step first ...
    v16bf a  = load_frag_a(abase + kb, DMODEL);
    v16bf b0 = load_frag_b(bbase + (size_t)0  * DMODEL + kb, DMODEL);
    v16bf b1 = load_frag_b(bbase + (size_t)16 * DMODEL + kb, DMODEL);
    v16bf b2 = load_frag_b(bbase + (size_t)32 * DMODEL + kb, DMODEL);
    v16bf b3 = load_frag_b(bbase + (size_t)48 * DMODEL + kb, DMODEL);
    // ... then the 4 matrix ops
    c0 = wmma_bf16(a, b0, c0);
    c1 = wmma_bf16(a, b1, c1);
    c2 = wmma_bf16(a, b2, c2);
    c3 = wmma_bf16(a, b3, c3);
  }

  const int rbase = tm * 16 + ((lane < 16) ? 0 : 8);
  v8f acc[4] = {c0, c1, c2, c3};
#pragma unroll
  for (int j = 0; j < 4; ++j) {
    const int col = tn8 * 64 + j * 16 + (lane & 15);
    if (mat < 2) {
      __bf16* O = (mat == 0) ? Qb : Kb;
#pragma unroll
      for (int v = 0; v < 8; ++v)
        O[(size_t)(rbase + v) * DMODEL + col] = (__bf16)acc[j][v];
    } else {
      const int h = col >> 6, hd = col & 63;
#pragma unroll
      for (int v = 0; v < 8; ++v) {
        int t  = rbase + v;
        int bb = t >> 11;       // batch
        int nt = t & (SEQ - 1); // token within sequence
        Vt[(((size_t)(bb * NHEAD + h)) * HDIM + hd) * SEQ + nt] = (__bf16)acc[j][v];
      }
    }
  }
}

// ---------------------------------------------------------------------------
// Kernel 4: flash attention + residual.
// Block = 128 threads (4 waves) all on the same (batch, head); wave w handles
// query tile qg*4+w. The 32-token K/V chunk is shared by the whole block and
// double-buffered in LDS via CDNA5 async global->LDS DMA (ASYNCcnt), issued
// cooperatively by all 128 threads while the current chunk's WMMAs run.
// LDS: K 2x4KB + V 2x4KB + P-stage 4x1KB = 20 KB/block.
// ---------------------------------------------------------------------------
__global__ __launch_bounds__(128) void attn_kernel(const __bf16* __restrict__ Qb,
                                                   const __bf16* __restrict__ Kb,
                                                   const __bf16* __restrict__ Vt,
                                                   const float* __restrict__ xn,
                                                   float* __restrict__ out) {
  __shared__ __bf16 kbuf[2][32 * HDIM];   // [token][hd], ld=64
  __shared__ __bf16 vbuf[2][HDIM * 32];   // [hd][token], ld=32
  __shared__ __bf16 pstage[4][16 * 32];   // per-wave P tile, ld=32

  const int tid  = threadIdx.x;
  const int wave = tid >> 5;
  const int lane = tid & 31;
  const int qg = blockIdx.x & 31;         // 32 query groups of 4 tiles
  const int h  = (blockIdx.x >> 5) & 7;
  const int b  = blockIdx.x >> 8;
  const int qt = qg * 4 + wave;
  const size_t q0 = (size_t)b * SEQ + (size_t)qt * 16;  // first global token row
  const float scale = 0.044194173824159216f;            // 1/sqrt(512)

  const __bf16* Khead = Kb + (size_t)b * SEQ * DMODEL + h * HDIM;
  const __bf16* Vhead = Vt + ((size_t)(b * NHEAD + h)) * HDIM * SEQ;

  // Q fragments live in registers for the whole KV sweep
  const v16bf aq0 = load_frag_a(Qb + q0 * DMODEL + h * HDIM + 0,  DMODEL);
  const v16bf aq1 = load_frag_a(Qb + q0 * DMODEL + h * HDIM + 32, DMODEL);

  float m[8], l[8];
  v8f o0 = {}, o1 = {}, o2 = {}, o3 = {};
#pragma unroll
  for (int v = 0; v < 8; ++v) { m[v] = -INFINITY; l[v] = 0.0f; }

  const int cc    = lane & 15;
  const int rbase = (lane < 16) ? 0 : 8;

  // cooperative 32-token KV chunk stage: K = 256 16B chunks, V = 256 16B chunks
  auto stage_chunk = [&](int j0, int buf) {
#pragma unroll
    for (int r = 0; r < 2; ++r) {
      int ck = tid + r * 128;                        // K chunk id 0..255
      int krow = ck >> 3, ksub = ck & 7;             // row=token, 8x16B per row
      copy16_g2l(Khead + (size_t)(j0 + krow) * DMODEL + ksub * 8,
                 &kbuf[buf][krow * HDIM + ksub * 8]);
      int cv = tid + r * 128;                        // V chunk id 0..255
      int vrow = cv >> 2, vsub = cv & 3;             // row=hd, 4x16B per row
      copy16_g2l(Vhead + (size_t)vrow * SEQ + j0 + vsub * 8,
                 &vbuf[buf][vrow * 32 + vsub * 8]);
    }
  };

  stage_chunk(0, 0);
  wait_asynccnt0();
  __syncthreads();

  int cur = 0;
  for (int j0 = 0; j0 < SEQ; j0 += 32) {
    if (j0 + 32 < SEQ) stage_chunk(j0 + 32, cur ^ 1);   // overlap next DMA

    // scores from LDS: issue all 4 K-fragment loads, then the 4 WMMAs
    v16bf bk00 = load_frag_b(&kbuf[cur][0],              HDIM);
    v16bf bk01 = load_frag_b(&kbuf[cur][32],             HDIM);
    v16bf bk10 = load_frag_b(&kbuf[cur][16 * HDIM],      HDIM);
    v16bf bk11 = load_frag_b(&kbuf[cur][16 * HDIM + 32], HDIM);
    v8f s0 = {}, s1 = {};
    s0 = wmma_bf16(aq0, bk00, s0);
    s0 = wmma_bf16(aq1, bk01, s0);
    s1 = wmma_bf16(aq0, bk10, s1);
    s1 = wmma_bf16(aq1, bk11, s1);

    float p0[8], p1[8], alpha[8];
#pragma unroll
    for (int v = 0; v < 8; ++v) {
      float a0 = s0[v] * scale, a1 = s1[v] * scale;
      float mx = fmaxf(a0, a1);
#pragma unroll
      for (int msk = 1; msk <= 8; msk <<= 1) mx = fmaxf(mx, __shfl_xor(mx, msk, 32));
      float mn = fmaxf(m[v], mx);
      alpha[v] = __expf(m[v] - mn);
      m[v] = mn;
      p0[v] = __expf(a0 - mn);
      p1[v] = __expf(a1 - mn);
      float ps = p0[v] + p1[v];
#pragma unroll
      for (int msk = 1; msk <= 8; msk <<= 1) ps += __shfl_xor(ps, msk, 32);
      l[v] = l[v] * alpha[v] + ps;
    }

    // stage P (16x32 bf16) to LDS: D-layout -> row-major (per-wave region)
#pragma unroll
    for (int v = 0; v < 8; ++v) {
      pstage[wave][(rbase + v) * 32 + cc]      = (__bf16)p0[v];
      pstage[wave][(rbase + v) * 32 + cc + 16] = (__bf16)p1[v];
    }
    __syncthreads();
    v16bf pf = load_frag_a(&pstage[wave][0], 32);

    // issue all 4 V-fragment loads, rescale context, then the 4 WMMAs
    v16bf bv0 = load_frag_b(&vbuf[cur][0 * 16 * 32], 32);
    v16bf bv1 = load_frag_b(&vbuf[cur][1 * 16 * 32], 32);
    v16bf bv2 = load_frag_b(&vbuf[cur][2 * 16 * 32], 32);
    v16bf bv3 = load_frag_b(&vbuf[cur][3 * 16 * 32], 32);
#pragma unroll
    for (int v = 0; v < 8; ++v) {
      o0[v] *= alpha[v]; o1[v] *= alpha[v]; o2[v] *= alpha[v]; o3[v] *= alpha[v];
    }
    o0 = wmma_bf16(pf, bv0, o0);
    o1 = wmma_bf16(pf, bv1, o1);
    o2 = wmma_bf16(pf, bv2, o2);
    o3 = wmma_bf16(pf, bv3, o3);

    wait_asynccnt0();     // next chunk's DMA landed
    __syncthreads();      // everyone done with cur buffers + pstage
    cur ^= 1;
  }

  // epilogue: divide by softmax sum, add residual xn, store fp32
#pragma unroll
  for (int v = 0; v < 8; ++v) {
    float inv = 1.0f / l[v];
    size_t base = (q0 + rbase + v) * DMODEL + h * HDIM;
    out[base + cc]      = o0[v] * inv + xn[base + cc];
    out[base + 16 + cc] = o1[v] * inv + xn[base + 16 + cc];
    out[base + 32 + cc] = o2[v] * inv + xn[base + 32 + cc];
    out[base + 48 + cc] = o3[v] * inv + xn[base + 48 + cc];
  }
}

// ---------------------------------------------------------------------------
// Host-side launch
// ---------------------------------------------------------------------------
extern "C" void kernel_launch(void* const* d_in, const int* in_sizes, int n_in,
                              void* d_out, int out_size, void* d_ws, size_t ws_size,
                              hipStream_t stream) {
  const float* x     = (const float*)d_in[0];
  const float* Wq    = (const float*)d_in[1];
  const float* Wk    = (const float*)d_in[2];
  const float* Wv    = (const float*)d_in[3];
  const float* gamma = (const float*)d_in[4];
  const float* beta  = (const float*)d_in[5];
  float* out = (float*)d_out;

  // workspace layout (bytes): ~50 MB total
  char* ws = (char*)d_ws;
  const size_t XN_F32  = (size_t)NTOK * DMODEL * sizeof(float);
  const size_t XN_BF   = (size_t)NTOK * DMODEL * sizeof(__bf16);
  const size_t W_BF    = (size_t)DMODEL * DMODEL * sizeof(__bf16);
  const size_t QKV_BF  = (size_t)NTOK * DMODEL * sizeof(__bf16);

  float*  xn   = (float*)(ws);
  __bf16* xnbf = (__bf16*)(ws + XN_F32);
  __bf16* Wqb  = (__bf16*)(ws + XN_F32 + XN_BF);
  __bf16* Wkb  = (__bf16*)(ws + XN_F32 + XN_BF + W_BF);
  __bf16* Wvb  = (__bf16*)(ws + XN_F32 + XN_BF + 2 * W_BF);
  __bf16* Qb   = (__bf16*)(ws + XN_F32 + XN_BF + 3 * W_BF);
  __bf16* Kb   = (__bf16*)(ws + XN_F32 + XN_BF + 3 * W_BF + QKV_BF);
  __bf16* Vt   = (__bf16*)(ws + XN_F32 + XN_BF + 3 * W_BF + 2 * QKV_BF);

  const int wn = DMODEL * DMODEL;  // 262144
  cvt_bf16_kernel<<<wn / 256, 256, 0, stream>>>(Wq, Wqb, wn);
  cvt_bf16_kernel<<<wn / 256, 256, 0, stream>>>(Wk, Wkb, wn);
  cvt_bf16_kernel<<<wn / 256, 256, 0, stream>>>(Wv, Wvb, wn);

  ln_kernel<<<NTOK / 8, 256, 0, stream>>>(x, gamma, beta, xn, xnbf);

  // 3 matrices * 512 m-tiles * 8 strip-tiles = 12288 waves, 8 waves/block
  qkv_kernel<<<12288 / 8, 256, 0, stream>>>(xnbf, Wqb, Wkb, Wvb, Qb, Kb, Vt);

  // 4 batch * 8 heads * 32 query groups = 1024 blocks of 128 threads
  attn_kernel<<<1024, 128, 0, stream>>>(Qb, Kb, Vt, xn, out);
}
